// RNNModel_51969104281663
// MI455X (gfx1250) — matrix operand order repaired and checked by reading, not compiled, not run on hardware
//
#include <hip/hip_runtime.h>
#include <hip/hip_bf16.h>

// ---------------------------------------------------------------------------
// RNN  (T=512, B=64, in=1024, hid=2048, out=1024)  for gfx1250 (MI455X)
//   Phase 0: fp32 -> bf16 converts (X, W_ih, W_hh, W_out), zero h0 + sync ctr
//   Phase 1: Z = X @ W_ih^T            64x32-per-wave WMMA GEMM, fp32 out
//   Phase 2: ONE persistent kernel, 512 steps:
//              H[t+1] = relu(Z_t + H[t] @ W_hh^T)
//            K split across 2 waves/tile (LDS reduce), grid-wide atomic
//            barrier between steps; Z bias prefetched under the GEMM.
//   Phase 3: out[b][t][:] = H[1+t] @ W_out^T   64x32-per-wave WMMA GEMM,
//            non-temporal stores (output is write-once, keep L2 clean).
// All matmuls use v_wmma_f32_16x16x32_bf16 (wave32, f32 accumulate).
// ---------------------------------------------------------------------------

typedef __attribute__((ext_vector_type(16))) __bf16 v16bf;
typedef __attribute__((ext_vector_type(8)))  float  v8f;

#define T_STEPS 512
#define BATCH   64
#define NIN     1024
#define NHID    2048
#define NOUT    1024
#define RNN_BLOCKS 128   // persistent-kernel grid (must be co-resident)

// ---- WMMA fragment loaders (ISA 7.12.2 layouts, wave32) -------------------
// A 16x32 bf16: lane m = lane&15; khalf = (lane>>4)*8
//   elems 0..7  -> k = k0 + khalf + i         (contiguous, 16B)
//   elems 8..15 -> k = k0 + 16 + khalf + i-8  (contiguous, 16B)
__device__ __forceinline__ v16bf ld_frag_a(const __bf16* base, int ld,
                                           int row0, int k0, int lane) {
  const int m    = lane & 15;
  const int koff = (lane >> 4) << 3;
  const __bf16* p = base + (size_t)(row0 + m) * ld + (k0 + koff);
  union { uint4 u[2]; v16bf v; } u;
  u.u[0] = *(const uint4*)(p);        // global_load_b128
  u.u[1] = *(const uint4*)(p + 16);   // global_load_b128
  return u.v;
}

// B 32x16 bf16 (h @ W^T => B[k][n] = W[n][k], W row-major along k):
//   lane n = lane&15; koff = (lane>>4)*8; elems 0..7 -> k = k0+koff+i, elems 8..15 -> k = k0+16+koff+i-8
__device__ __forceinline__ v16bf ld_frag_b(const __bf16* base, int ld,
                                           int col0, int k0, int lane) {
  const int n    = lane & 15;
  const int koff = (lane >> 4) << 3;
  const __bf16* p = base + (size_t)(col0 + n) * ld + (k0 + koff);
  union { uint4 u[2]; v16bf v; } u;
  u.u[0] = *(const uint4*)(p);
  u.u[1] = *(const uint4*)(p + 16);
  return u.v;
}
// C/D 16x16 f32: lane n = lane&15; VGPR r -> m = r + (lane>=16 ? 8 : 0)

#define WMMA_BF16(A, B, C) \
  __builtin_amdgcn_wmma_f32_16x16x32_bf16(false, (A), false, (B), (short)0, (C), false, false)

// ---- Phase 0 helpers ------------------------------------------------------
__global__ __launch_bounds__(256) void k_f32_to_bf16(const float* __restrict__ in,
                                                     __bf16* __restrict__ out, int n) {
  int i = blockIdx.x * 256 + threadIdx.x;
  if (i < n) out[i] = (__bf16)in[i];
}

__global__ __launch_bounds__(256) void k_zero_bf16(__bf16* __restrict__ p, int n) {
  int i = blockIdx.x * 256 + threadIdx.x;
  if (i < n) p[i] = (__bf16)0.0f;
}

__global__ __launch_bounds__(64) void k_zero_u32(unsigned* __restrict__ p, int n) {
  int i = threadIdx.x;
  if (i < n) p[i] = 0u;
}

// ---- Blocked GEMM body: 64x32 output per wave (4x2 tiles, 8 WMMA chains) --
__device__ __forceinline__ void gemm_64x32(const __bf16* __restrict__ A, int lda,
                                           const __bf16* __restrict__ B, int ldb,
                                           int m0, int n0, int K, int lane,
                                           v8f acc[4][2]) {
  for (int k0 = 0; k0 < K; k0 += 32) {
    v16bf af[4], bf[2];
    #pragma unroll
    for (int i = 0; i < 4; ++i) af[i] = ld_frag_a(A, lda, m0 + 16 * i, k0, lane);
    #pragma unroll
    for (int j = 0; j < 2; ++j) bf[j] = ld_frag_b(B, ldb, n0 + 16 * j, k0, lane);
    #pragma unroll
    for (int i = 0; i < 4; ++i)
      #pragma unroll
      for (int j = 0; j < 2; ++j)
        acc[i][j] = WMMA_BF16(af[i], bf[j], acc[i][j]);
  }
}

// ---- Phase 1: Z[m][n] = sum_k X[m][k]*Wih[n][k];  M=32768 N=2048 K=1024 ---
// waves = (M/64)*(N/32) = 512*64 = 32768 -> 4096 blocks
__global__ __launch_bounds__(256) void k_in_proj(const __bf16* __restrict__ X,
                                                 const __bf16* __restrict__ W,
                                                 float* __restrict__ Z) {
  const int lane = threadIdx.x & 31;
  const int wid  = blockIdx.x * 8 + (threadIdx.x >> 5);
  const int n0   = (wid & 63) << 5;           // 64 wave-cols * 32
  const int m0   = (wid >> 6) << 6;           // 512 wave-rows * 64
  v8f acc[4][2];
  #pragma unroll
  for (int i = 0; i < 4; ++i)
    #pragma unroll
    for (int j = 0; j < 2; ++j) acc[i][j] = (v8f){};
  gemm_64x32(X, NIN, W, NIN, m0, n0, NIN, lane, acc);
  const int nl = lane & 15, moff = (lane >> 4) << 3;
  #pragma unroll
  for (int i = 0; i < 4; ++i)
    #pragma unroll
    for (int j = 0; j < 2; ++j)
      #pragma unroll
      for (int r = 0; r < 8; ++r)
        Z[(size_t)(m0 + 16 * i + moff + r) * NHID + (n0 + 16 * j + nl)] = acc[i][j][r];
}

// ---- Phase 2: persistent recurrence kernel --------------------------------
// 128 blocks x 8 waves = 1024 waves = 512 tiles (4m x 128n) x 2 K-halves.
// Waves 0..3 of a block do K[0,1024), waves 4..7 K[1024,2048); halves are
// combined through LDS, biased with prefetched Z, relu'd, stored bf16.
// Steps separated by a device-scope atomic-counter barrier.
__global__ __launch_bounds__(256) void k_rnn_persistent(const __bf16* __restrict__ Whh,
                                                        const float*  __restrict__ Z,
                                                        __bf16* __restrict__ Hbuf,
                                                        unsigned* __restrict__ syncCnt) {
  const int lane  = threadIdx.x & 31;
  const int w     = threadIdx.x >> 5;          // 0..7
  const int tsub  = w & 3;                     // tile slot in block
  const int khalf = w >> 2;                    // 0 or 1
  const int tile  = blockIdx.x * 4 + tsub;     // 0..511
  const int n0    = (tile & 127) << 4;
  const int m0    = (tile >> 7) << 4;
  const int kbase = khalf * (NHID / 2);
  const int nl    = n0 + (lane & 15);
  const int moff  = m0 + ((lane >> 4) << 3);

  __shared__ float red[4][32][8];              // per-tile partial sums (4 KB)

  for (int t = 0; t < T_STEPS; ++t) {
    const __bf16* Hprev = Hbuf + (size_t)t * BATCH * NHID;
    __bf16*       Hnew  = Hbuf + (size_t)(t + 1) * BATCH * NHID;
    const float*  Zt    = Z    + (size_t)t * BATCH * NHID;

    // Prefetch the Z bias BEFORE the GEMM so its L2 latency hides under the
    // 16-iteration WMMA loop instead of extending the serial critical path.
    float zbias[8];
    if (khalf == 0) {
      #pragma unroll
      for (int r = 0; r < 8; ++r)
        zbias[r] = Zt[(size_t)(moff + r) * NHID + nl];
    }

    v8f c0 = {}, c1 = {};
    #pragma unroll 2
    for (int k0 = kbase; k0 < kbase + NHID / 2; k0 += 64) {
      v16bf a0 = ld_frag_a(Hprev, NHID, m0, k0,      lane);
      v16bf b0 = ld_frag_b(Whh,   NHID, n0, k0,      lane);
      v16bf a1 = ld_frag_a(Hprev, NHID, m0, k0 + 32, lane);
      v16bf b1 = ld_frag_b(Whh,   NHID, n0, k0 + 32, lane);
      c0 = WMMA_BF16(a0, b0, c0);
      c1 = WMMA_BF16(a1, b1, c1);
    }

    if (khalf == 1) {                          // publish upper-K partial
      #pragma unroll
      for (int r = 0; r < 8; ++r) red[tsub][lane][r] = c0[r] + c1[r];
    }
    __syncthreads();
    if (khalf == 0) {                          // combine + bias + relu + store
      #pragma unroll
      for (int r = 0; r < 8; ++r) {
        float v = c0[r] + c1[r] + red[tsub][lane][r] + zbias[r];
        v = v > 0.0f ? v : 0.0f;
        Hnew[(size_t)(moff + r) * NHID + nl] = (__bf16)v;
      }
    }

    // ---- device-wide step barrier ----
    __threadfence();                           // release H stores to device
    __syncthreads();                           // (also guards LDS reuse)
    if (threadIdx.x == 0) {
      atomicAdd(syncCnt, 1u);
      const unsigned target = (unsigned)gridDim.x * (unsigned)(t + 1);
      while (__hip_atomic_load(syncCnt, __ATOMIC_ACQUIRE,
                               __HIP_MEMORY_SCOPE_AGENT) < target)
        __builtin_amdgcn_s_sleep(1);
    }
    __syncthreads();
  }
}

// ---- Phase 3: out[b][t][n] = sum_k H[t*64+b][k]*Wout[n][k] ----------------
// M=32768 N=1024 K=2048; waves = 512*32 = 16384 -> 2048 blocks
// Output is streamed once -> non-temporal stores keep it out of L2.
__global__ __launch_bounds__(256) void k_out_proj(const __bf16* __restrict__ H,
                                                  const __bf16* __restrict__ W,
                                                  float* __restrict__ Out) {
  const int lane = threadIdx.x & 31;
  const int wid  = blockIdx.x * 8 + (threadIdx.x >> 5);
  const int n0   = (wid & 31) << 5;           // 32 wave-cols * 32
  const int m0   = (wid >> 5) << 6;           // 512 wave-rows * 64
  v8f acc[4][2];
  #pragma unroll
  for (int i = 0; i < 4; ++i)
    #pragma unroll
    for (int j = 0; j < 2; ++j) acc[i][j] = (v8f){};
  gemm_64x32(H, NHID, W, NHID, m0, n0, NHID, lane, acc);
  const int nl = lane & 15, moff = (lane >> 4) << 3;
  #pragma unroll
  for (int i = 0; i < 4; ++i)
    #pragma unroll
    for (int j = 0; j < 2; ++j)
      #pragma unroll
      for (int r = 0; r < 8; ++r) {
        const int m = m0 + 16 * i + moff + r;  // m = t*64 + b
        const int t = m >> 6, b = m & 63;
        __builtin_nontemporal_store(acc[i][j][r],
            Out + ((size_t)b * T_STEPS + t) * NOUT + (n0 + 16 * j + nl));
      }
}

// ---------------------------------------------------------------------------
extern "C" void kernel_launch(void* const* d_in, const int* in_sizes, int n_in,
                              void* d_out, int out_size, void* d_ws, size_t ws_size,
                              hipStream_t stream) {
  const float* xs   = (const float*)d_in[0];  // [T,B,NIN]
  const float* Wih  = (const float*)d_in[1];  // [NHID,NIN]
  const float* Whh  = (const float*)d_in[2];  // [NHID,NHID]
  const float* Wout = (const float*)d_in[3];  // [NOUT,NHID]

  char* ws = (char*)d_ws;
  size_t off = 0;
  __bf16*  Xbf   = (__bf16*)(ws + off);  off += (size_t)T_STEPS * BATCH * NIN  * 2;
  __bf16*  WihB  = (__bf16*)(ws + off);  off += (size_t)NHID * NIN  * 2;
  __bf16*  WhhB  = (__bf16*)(ws + off);  off += (size_t)NHID * NHID * 2;
  __bf16*  WoutB = (__bf16*)(ws + off);  off += (size_t)NOUT * NHID * 2;
  float*   Z     = (float*) (ws + off);  off += (size_t)T_STEPS * BATCH * NHID * 4;
  __bf16*  Hbuf  = (__bf16*)(ws + off);  off += (size_t)(T_STEPS + 1) * BATCH * NHID * 2;
  unsigned* syncCnt = (unsigned*)(ws + off); off += 256;
  (void)ws_size; (void)in_sizes; (void)n_in; (void)out_size;

  // Phase 0: precision converts + h0 = 0 + barrier counter = 0
  {
    int nX = T_STEPS * BATCH * NIN;
    k_f32_to_bf16<<<(nX + 255) / 256, 256, 0, stream>>>(xs, Xbf, nX);
    int nW1 = NHID * NIN;
    k_f32_to_bf16<<<(nW1 + 255) / 256, 256, 0, stream>>>(Wih, WihB, nW1);
    int nW2 = NHID * NHID;
    k_f32_to_bf16<<<(nW2 + 255) / 256, 256, 0, stream>>>(Whh, WhhB, nW2);
    int nW3 = NOUT * NHID;
    k_f32_to_bf16<<<(nW3 + 255) / 256, 256, 0, stream>>>(Wout, WoutB, nW3);
    int nH0 = BATCH * NHID;
    k_zero_bf16<<<(nH0 + 255) / 256, 256, 0, stream>>>(Hbuf, nH0);
    k_zero_u32<<<1, 64, 0, stream>>>(syncCnt, 64);
  }

  // Phase 1: Z = X @ Wih^T
  k_in_proj<<<4096, 256, 0, stream>>>(Xbf, WihB, Z);

  // Phase 2: full recurrence in a single persistent launch
  k_rnn_persistent<<<RNN_BLOCKS, 256, 0, stream>>>(WhhB, Z, Hbuf, syncCnt);

  // Phase 3: out = H @ Wout^T, scattered into [B,T,NOUT]
  k_out_proj<<<2048, 256, 0, stream>>>(Hbuf + (size_t)BATCH * NHID, WoutB,
                                       (float*)d_out);
}
